// PerformerEncoderAttention_23811298689621
// MI455X (gfx1250) — matrix operand order repaired and checked by reading, not compiled, not run on hardware
//
#include <hip/hip_runtime.h>

// ---------------------------------------------------------------------------
// FAVOR+ causal attention for MI455X (gfx1250), wave32 + WMMA f32_16x16x32_f16
// B=4 L=2048 H=8 D=64 M=256.  Pipeline:
//   init        : proj f32->f16, V f32->f16 transposed to [B*H][L][D], kmax=-inf
//   feat mode 0 : qp = ratio*(exp(dash - diag - rowmax)+eps)   -> f16 [B*H][L][M]
//   feat mode 1 : key pass 1: global max(dash) per (b,h) via atomic float max
//   feat mode 2 : kp = ratio*(exp(dash - diag - kmax[bh])+eps) -> f16 [B*H][L][M]
//   causal      : 32 blocks (one per b,h), chunked scan C=32, P state in WMMA
//                 accumulators (8 waves x 8 tiles of 16x16 f32).  Chunk tiles
//                 staged with GLOBAL_LOAD_ASYNC_TO_LDS_B128 (ASYNCcnt path).
// Workspace layout (needs ~80MB): projb@0(32KB) kmax@32KB vb@64KB(8MB)
//                                 qp@16MB(32MB) kp@48MB(32MB)
// ---------------------------------------------------------------------------

typedef __attribute__((ext_vector_type(16))) _Float16 v16h;
typedef __attribute__((ext_vector_type(8)))  float    v8f;

#if defined(__has_builtin)
#if __has_builtin(__builtin_amdgcn_global_load_async_to_lds_b128)
#define HAVE_ASYNC_B128 1
#endif
#endif
#ifndef HAVE_ASYNC_B128
#define HAVE_ASYNC_B128 0
#endif

#if HAVE_ASYNC_B128
// Param type per clang diagnostic: pointer to int4 vector; src in AS1
// (__device__/global), dst in AS3 (LDS).
typedef int v4i __attribute__((ext_vector_type(4)));
typedef __attribute__((address_space(1))) v4i* gptr4;
typedef __attribute__((address_space(3))) v4i* lptr4;
#endif

#define DEV static __device__ __forceinline__

constexpr int   Bb = 4, Ls = 2048, Hh = 8, Dd = 64, Mm = 256;
constexpr float NORMZ = 0.35355339059327373f;  // 1/64^(1/4)
constexpr float RATIO = 0.0625f;               // 1/sqrt(256)
constexpr float KEPS  = 1e-4f;
constexpr float NSTAB = 1e-6f;
constexpr float NEGBIG = -3.3e38f;

DEV v8f zero8() {
  v8f z;
#pragma unroll
  for (int i = 0; i < 8; ++i) z[i] = 0.f;
  return z;
}

DEV v8f wmma_f16(v16h a, v16h b, v8f c) {
  return __builtin_amdgcn_wmma_f32_16x16x32_f16(false, a, false, b,
                                                (short)0, c, false, false);
}

// A fragment (16x32 f16): lane<16 row=lane K=0..15;
// lane>=16 row=lane-16 K=16..31. p is row-major with leading dim ld.
DEV v16h frag_A(const _Float16* p, int ld, int lane) {
  const int m = lane & 15;
  const int kb = (lane & 16) ? 16 : 0;
  const _Float16* row = p + m * ld;
  v16h a;
#pragma unroll
  for (int e = 0; e < 8; ++e) a[e] = row[kb + e];
#pragma unroll
  for (int e = 0; e < 8; ++e) a[8 + e] = row[kb + 8 + e];
  return a;
}

// B fragment (32x16 f16): lane<16 col n=lane K=0..15; lane>=16 K=16..31.
// Source stored K-major: p[k*ld + n].
DEV v16h frag_B_KN(const _Float16* p, int ld, int lane) {
  const int n = lane & 15;
  const int kb = (lane & 16) ? 16 : 0;
  v16h b;
#pragma unroll
  for (int e = 0; e < 16; ++e) b[e] = p[(kb + e) * ld + n];
  return b;
}

// B fragment with source stored N-major: B[k][n] = p[n*ld + k] (contiguous K).
DEV v16h frag_B_NK(const _Float16* p, int ld, int lane) {
  const int n = lane & 15;
  const int kb = (lane & 16) ? 16 : 0;
  const _Float16* row = p + n * ld;
  v16h b;
#pragma unroll
  for (int e = 0; e < 16; ++e) b[e] = row[kb + e];
  return b;
}

// C/D 16x16 f32: VGPR r, lanes 0-15: (m=r, n=lane); lanes 16-31: (m=8+r).
DEV void store_C_f(float* p, int ld, int lane, v8f c) {
  const int n = lane & 15, mo = (lane & 16) ? 8 : 0;
#pragma unroll
  for (int r = 0; r < 8; ++r) p[(mo + r) * ld + n] = c[r];
}
DEV void store_C_h(_Float16* p, int ld, int lane, v8f c) {
  const int n = lane & 15, mo = (lane & 16) ? 8 : 0;
#pragma unroll
  for (int r = 0; r < 8; ++r) p[(mo + r) * ld + n] = (_Float16)c[r];
}

DEV void atomicMaxF(float* addr, float val) {
  if (val >= 0.f) atomicMax((int*)addr, __float_as_int(val));
  else            atomicMin((unsigned int*)addr, __float_as_uint(val));
}

// ---------------------------------------------------------------------------
__global__ __launch_bounds__(256) void favor_init_kernel(
    const float* __restrict__ proj, const float* __restrict__ v,
    _Float16* __restrict__ projb, _Float16* __restrict__ vb,
    float* __restrict__ kmax) {
  const int tid = blockIdx.x * 256 + threadIdx.x;
  if (tid < Bb * Hh) kmax[tid] = NEGBIG;
  if (tid < Mm * Dd) projb[tid] = (_Float16)proj[tid];
  const int total = Bb * Hh * Ls * Dd;  // 4M, [B*H][L][D] layout
  for (int o = tid; o < total; o += gridDim.x * 256) {
    const int d = o & 63, l = (o >> 6) & 2047, h = (o >> 17) & 7, b = o >> 20;
    vb[o] = (_Float16)v[(((size_t)(b * Ls + l)) * Hh + h) * Dd + d];
  }
}

// ---------------------------------------------------------------------------
// dash = (NORMZ*X) @ proj^T for a 16-row tile; modes as documented above.
__global__ __launch_bounds__(128) void favor_feat_kernel(
    const float* __restrict__ x, const _Float16* __restrict__ projb,
    _Float16* __restrict__ outp, float* __restrict__ kmax, const int mode) {
  __shared__ _Float16 Xs[16 * 64];
  __shared__ float diag[16];
  __shared__ float dashs[16 * 256];
  __shared__ float red[16 * 8];
  __shared__ float rmax[16];

  const int t  = threadIdx.x;
  const int r0 = blockIdx.x * 16;
  {
    const int i = t >> 3, d0 = (t & 7) * 8;
    const float* xr = x + (size_t)(r0 + i) * Dd + d0;
    float ssq = 0.f;
#pragma unroll
    for (int e = 0; e < 8; ++e) {
      const float vv = xr[e];
      Xs[i * Dd + d0 + e] = (_Float16)(vv * NORMZ);
      ssq += vv * vv;
    }
    red[i * 8 + (t & 7)] = ssq;
  }
  __syncthreads();
  if (t < 16) {
    float s = 0.f;
#pragma unroll
    for (int j = 0; j < 8; ++j) s += red[t * 8 + j];
    diag[t] = s * 0.0625f;  // 0.5 * NORMZ^2
  }
  __syncthreads();

  const int wave = t >> 5, lane = t & 31;
  v8f acc[4];
#pragma unroll
  for (int nt = 0; nt < 4; ++nt) acc[nt] = zero8();
#pragma unroll
  for (int kk = 0; kk < 2; ++kk) {
    const v16h a = frag_A(Xs + kk * 32, Dd, lane);
#pragma unroll
    for (int nt = 0; nt < 4; ++nt) {
      const int n0 = wave * 64 + nt * 16;
      const v16h bb = frag_B_NK(projb + (size_t)n0 * Dd + kk * 32, Dd, lane);
      acc[nt] = wmma_f16(a, bb, acc[nt]);
    }
  }
#pragma unroll
  for (int nt = 0; nt < 4; ++nt)
    store_C_f(dashs + wave * 64 + nt * 16, Mm, lane, acc[nt]);
  __syncthreads();

  if (t < 16) {
    float mx = NEGBIG;
    for (int j = 0; j < Mm; ++j) mx = fmaxf(mx, dashs[t * Mm + j]);
    rmax[t] = mx;
  }
  __syncthreads();

  if (mode == 1) {
    if (t < 16) {
      const int rr = r0 + t;
      atomicMaxF(&kmax[(rr >> 14) * Hh + (rr & 7)], rmax[t]);
    }
    return;
  }

  const int i = t >> 3;
  const int r = r0 + i;
  const int bi = r >> 14, lh = r & 16383, li = lh >> 3, hi = lh & 7;
  const float stab = (mode == 0) ? rmax[i] : kmax[bi * Hh + hi];
  const float dg = diag[i];
  _Float16* orow = outp + (((size_t)(bi * Hh + hi)) * Ls + li) * Mm;
  const int m0 = (t & 7) * 32;
#pragma unroll
  for (int c = 0; c < 32; ++c) {
    const int m = m0 + c;
    orow[m] = (_Float16)(RATIO * (__expf(dashs[i * Mm + m] - dg - stab) + KEPS));
  }
}

// ---------------------------------------------------------------------------
// Chunked causal scan. One block per (b,h), 8 waves. P[256][64] lives in WMMA
// accumulators: wave w owns M rows [32w,32w+32) as 2x4 tiles.
__global__ __launch_bounds__(256) void favor_causal_kernel(
    const _Float16* __restrict__ qp, const _Float16* __restrict__ kp,
    const _Float16* __restrict__ vb, float* __restrict__ out) {
  extern __shared__ char smem[];
  _Float16* Qs = (_Float16*)(smem);           // [32][256] 16KB
  _Float16* Ks = (_Float16*)(smem + 16384);   // [32][256] 16KB
  _Float16* Vs = (_Float16*)(smem + 32768);   // [32][64]   4KB
  _Float16* Pb = (_Float16*)(smem + 36864);   // [256][64] 32KB (f16 shadow of P)
  _Float16* Ab = (_Float16*)(smem + 69632);   // [32][32]   2KB (masked A)
  float* sv    = (float*)(smem + 71680);      // [256]      s state
  float* denqs = (float*)(smem + 72704);      // [32][4]
  float* den   = (float*)(smem + 73216);      // [32]

  const int tid = threadIdx.x, wave = tid >> 5, lane = tid & 31;
  const int bh = blockIdx.x, bi = bh >> 3, hi = bh & 7;
  const _Float16* qrow = qp + (size_t)bh * Ls * Mm;
  const _Float16* krow = kp + (size_t)bh * Ls * Mm;
  const _Float16* vrow = vb + (size_t)bh * Ls * Dd;

  v8f P[8];
#pragma unroll
  for (int i = 0; i < 8; ++i) P[i] = zero8();
  sv[tid] = 0.f;
  __syncthreads();

  for (int t = 0; t < Ls / 32; ++t) {
    const int l0 = t * 32;
    // ---- stage chunk tiles into LDS ----
    {
      const uint4* s0 = (const uint4*)(qrow + (size_t)l0 * Mm);
      const uint4* s1 = (const uint4*)(krow + (size_t)l0 * Mm);
      const uint4* s2 = (const uint4*)(vrow + (size_t)l0 * Dd);
      uint4* d0 = (uint4*)Qs; uint4* d1 = (uint4*)Ks; uint4* d2 = (uint4*)Vs;
#if HAVE_ASYNC_B128
      // CDNA5 async copy path: LDS written directly from memory hierarchy,
      // no VGPR staging; tracked by ASYNCcnt (CDNA5 ISA).
#pragma unroll
      for (int k = 0; k < 4; ++k) {
        __builtin_amdgcn_global_load_async_to_lds_b128(
            (gptr4)(s0 + tid + 256 * k), (lptr4)(d0 + tid + 256 * k), 0, 0);
        __builtin_amdgcn_global_load_async_to_lds_b128(
            (gptr4)(s1 + tid + 256 * k), (lptr4)(d1 + tid + 256 * k), 0, 0);
      }
      __builtin_amdgcn_global_load_async_to_lds_b128(
          (gptr4)(s2 + tid), (lptr4)(d2 + tid), 0, 0);
      asm volatile("s_wait_asynccnt 0x0" ::: "memory");
#else
#pragma unroll
      for (int k = 0; k < 4; ++k) {
        d0[tid + 256 * k] = s0[tid + 256 * k];
        d1[tid + 256 * k] = s1[tid + 256 * k];
      }
      d2[tid] = s2[tid];
#endif
    }
    if (t + 1 < Ls / 32) {  // stream next chunk toward L2/L0
      __builtin_prefetch((const void*)(qrow + (size_t)(l0 + 32) * Mm + tid * 32), 0, 1);
      __builtin_prefetch((const void*)(krow + (size_t)(l0 + 32) * Mm + tid * 32), 0, 1);
      if (tid < 64)
        __builtin_prefetch((const void*)(vrow + (size_t)(l0 + 32) * Dd + tid * 32), 0, 1);
    }
    __syncthreads();

    // ---- mirror P (prev state) to LDS as f16 B-operand ----
#pragma unroll
    for (int i = 0; i < 8; ++i) {
      const int mt = 2 * wave + (i >> 2), dt = i & 3;
      store_C_h(Pb + (mt * 16) * Dd + dt * 16, Dd, lane, P[i]);
    }
    __syncthreads();

    // ---- waves 0-3: A = Qp Kp^T (K=256) masked; waves 4-7: q.s_prev ----
    if (wave < 4) {
      const int ti = wave >> 1, tj = wave & 1;
      v8f a = zero8();
#pragma unroll
      for (int kk = 0; kk < 8; ++kk) {
        const v16h af = frag_A(Qs + ti * 16 * Mm + kk * 32, Mm, lane);
        const v16h bf = frag_B_NK(Ks + tj * 16 * Mm + kk * 32, Mm, lane);
        a = wmma_f16(af, bf, a);
      }
      const int n = lane & 15, mo = (lane & 16) ? 8 : 0;
      const int jg = tj * 16 + n;
#pragma unroll
      for (int r = 0; r < 8; ++r) {
        const int ig = ti * 16 + mo + r;
        Ab[ig * 32 + jg] = (_Float16)((jg <= ig) ? a[r] : 0.f);
      }
    } else {
      const int tt = tid - 128, i = tt & 31, part = tt >> 5;
      float acc = 0.f;
      const _Float16* qr = Qs + i * Mm + part * 64;
#pragma unroll 16
      for (int m = 0; m < 64; ++m) acc += (float)qr[m] * sv[part * 64 + m];
      denqs[i * 4 + part] = acc;
    }
    __syncthreads();

    // ---- finalize den[i] = q.s_prev + rowsum(tril A) + stabilizer ----
    if (tid < 32) {
      float dv = denqs[tid * 4] + denqs[tid * 4 + 1] + denqs[tid * 4 + 2] +
                 denqs[tid * 4 + 3];
      for (int j = 0; j <= tid; ++j) dv += (float)Ab[tid * 32 + j];
      dv += 2.f * NSTAB * ((fabsf(dv) <= NSTAB) ? 1.f : 0.f);
      den[tid] = dv;
    }
    __syncthreads();

    // ---- num = Qp @ P_prev (K=256) + tril(A) @ V (K=32); write output ----
    {
      const int rt = wave >> 2, ct = wave & 3;
      v8f acc = zero8();
#pragma unroll
      for (int kk = 0; kk < 8; ++kk) {
        const v16h af = frag_A(Qs + rt * 16 * Mm + kk * 32, Mm, lane);
        const v16h bf = frag_B_KN(Pb + (kk * 32) * Dd + ct * 16, Dd, lane);
        acc = wmma_f16(af, bf, acc);
      }
      {
        const v16h af = frag_A(Ab + rt * 16 * 32, 32, lane);
        const v16h bf = frag_B_KN(Vs + ct * 16, Dd, lane);
        acc = wmma_f16(af, bf, acc);
      }
      const int n = lane & 15, mo = (lane & 16) ? 8 : 0;
#pragma unroll
      for (int r = 0; r < 8; ++r) {
        const int i = rt * 16 + mo + r;
        out[(((size_t)bi * Ls + (l0 + i)) * Hh + hi) * Dd + ct * 16 + n] =
            acc[r] / den[i];
      }
    }

    // ---- state update: P += Kp^T V (K=32), s += sum_j Kp[j] ----
#pragma unroll
    for (int i = 0; i < 8; ++i) {
      const int mt = 2 * wave + (i >> 2), dt = i & 3;
      const int ml = lane & 15, kb = (lane & 16) ? 16 : 0;
      v16h af;
#pragma unroll
      for (int e = 0; e < 8; ++e) {
        af[e]     = Ks[(kb + e) * Mm + mt * 16 + ml];
        af[8 + e] = Ks[(kb + 8 + e) * Mm + mt * 16 + ml];
      }
      const v16h bf = frag_B_KN(Vs + dt * 16, Dd, lane);
      P[i] = wmma_f16(af, bf, P[i]);
    }
    {
      float acc = sv[tid];
#pragma unroll
      for (int j = 0; j < 32; ++j) acc += (float)Ks[j * Mm + tid];
      sv[tid] = acc;
    }
    __syncthreads();
  }
}

// ---------------------------------------------------------------------------
extern "C" void kernel_launch(void* const* d_in, const int* in_sizes, int n_in,
                              void* d_out, int out_size, void* d_ws,
                              size_t ws_size, hipStream_t stream) {
  (void)in_sizes; (void)n_in; (void)out_size; (void)ws_size;
  const float* q    = (const float*)d_in[0];
  const float* k    = (const float*)d_in[1];
  const float* v    = (const float*)d_in[2];
  const float* proj = (const float*)d_in[3];
  float* out = (float*)d_out;

  char* ws = (char*)d_ws;
  _Float16* projb = (_Float16*)(ws);                       // 32 KB
  float*    kmax  = (float*)(ws + 32768);                  // 128 B
  _Float16* vb    = (_Float16*)(ws + 65536);               // 8 MB
  _Float16* qpf   = (_Float16*)(ws + (16ull << 20));       // 32 MB
  _Float16* kpf   = (_Float16*)(ws + (48ull << 20));       // 32 MB

  const int featBlocks = (Bb * Ls * Hh) / 16;              // 4096
  favor_init_kernel<<<4096, 256, 0, stream>>>(proj, v, projb, vb, kmax);
  favor_feat_kernel<<<featBlocks, 128, 0, stream>>>(q, projb, qpf, kmax, 0);
  favor_feat_kernel<<<featBlocks, 128, 0, stream>>>(k, projb, kpf, kmax, 1);
  favor_feat_kernel<<<featBlocks, 128, 0, stream>>>(k, projb, kpf, kmax, 2);
  favor_causal_kernel<<<Bb * Hh, 256, 73344, stream>>>(qpf, kpf, vb, out);
}
